// EEG_LSM_28355374088682
// MI455X (gfx1250) — matrix operand order repaired and checked by reading, not compiled, not run on hardware
//
#include <hip/hip_runtime.h>

typedef __attribute__((ext_vector_type(16))) _Float16 v16h;
typedef __attribute__((ext_vector_type(2)))  _Float16 h2v;
typedef __attribute__((ext_vector_type(8)))  float    v8f;

#define TT    4000
#define CH    32          // timesteps staged per chunk
#define SPB   16          // samples per block
#define NTHR  128         // 4 waves (wave32)

// ---- LDS layout (in 32-bit words) ----
// xh: pre-packed half2 x-stage in WMMA A-fragment pair order:
//     [sample m][pair p (16, padded 17)][tc (32, padded 33)]
#define XH_PITCH_P 33
#define XH_PITCH_M (17*33)            // 561 (odd bank stride across samples)
#define XH_SZ      (16*XH_PITCH_M)    // 8976
#define W2T_OFF    XH_SZ              // [64][129]
#define WRT_OFF    (W2T_OFF + 64*129)
#define W3T_OFF    (WRT_OFF + 128*129)
#define INP1_OFF   (W3T_OFF + 128*129) // [16][68]
#define LDS_WORDS  (INP1_OFF + 16*68)  // 51344 words = 205376 bytes

// ---- single-instruction DPP8 xor-butterfly reduce steps (8-lane groups) ----
// v_max_num_f32 / v_min_i32 with a dpp8 source modifier fuse shuffle+op.
static __device__ __forceinline__ float maxred_x1(float v) {
    float r; asm("v_max_num_f32 %0, %1, %1 dpp8:[1,0,3,2,5,4,7,6]" : "=v"(r) : "v"(v)); return r;
}
static __device__ __forceinline__ float maxred_x2(float v) {
    float r; asm("v_max_num_f32 %0, %1, %1 dpp8:[2,3,0,1,6,7,4,5]" : "=v"(r) : "v"(v)); return r;
}
static __device__ __forceinline__ float maxred_x4(float v) {
    float r; asm("v_max_num_f32 %0, %1, %1 dpp8:[4,5,6,7,0,1,2,3]" : "=v"(r) : "v"(v)); return r;
}
static __device__ __forceinline__ int minred_x1(int v) {
    int r; asm("v_min_i32 %0, %1, %1 dpp8:[1,0,3,2,5,4,7,6]" : "=v"(r) : "v"(v)); return r;
}
static __device__ __forceinline__ int minred_x2(int v) {
    int r; asm("v_min_i32 %0, %1, %1 dpp8:[2,3,0,1,6,7,4,5]" : "=v"(r) : "v"(v)); return r;
}
static __device__ __forceinline__ int minred_x4(int v) {
    int r; asm("v_min_i32 %0, %1, %1 dpp8:[4,5,6,7,0,1,2,3]" : "=v"(r) : "v"(v)); return r;
}

// group-of-8 WTA: returns spike index (first max) or -1 if max < vth.
static __device__ __forceinline__ int wta_spike(float lmax, int lidx, float vth) {
    float gm = lmax;
    gm = maxred_x1(gm);
    gm = maxred_x2(gm);
    gm = maxred_x4(gm);
    int cand = (lmax == gm) ? lidx : 0x7FFFFFFF;
    cand = minred_x1(cand);
    cand = minred_x2(cand);
    cand = minred_x4(cand);
    return (gm >= vth) ? cand : -1;
}

static __device__ __forceinline__ unsigned pack_h2(float lo, float hi) {
    union { h2v h; unsigned u; } cv;
    cv.h[0] = (_Float16)lo;
    cv.h[1] = (_Float16)hi;
    return cv.u;
}

__global__ __launch_bounds__(NTHR, 1)
void eeg_lsm_scan_kernel(const float* __restrict__ x,   // [256][32][4000]
                         const float* __restrict__ W1,  // [64][32]
                         const float* __restrict__ W2,  // [128][64]
                         const float* __restrict__ Wr,  // [128][128]
                         const float* __restrict__ W3,  // [128][128]
                         float* __restrict__ out)       // [256][128]
{
    extern __shared__ float smem[];
    unsigned* xh   = (unsigned*)smem;     // packed half2 x-stage
    float*    W2T  = smem + W2T_OFF;      // W2T[k][n] = W2[n][k]
    float*    WrT  = smem + WRT_OFF;
    float*    W3T  = smem + W3T_OFF;
    float*    inp1 = smem + INP1_OFF;     // [16 samples][68 pad]

    const int tid     = threadIdx.x;
    const int lane    = tid & 31;
    const int wv      = tid >> 5;       // wave id -> layer-1 neuron tile
    const int halfsel = lane >> 4;
    const int lmod    = lane & 15;
    const int sb      = blockIdx.x * SPB;

    // ---- one-time: transpose weights into LDS (coalesced global reads) ----
    for (int i = tid; i < 128*64; i += NTHR) {
        int n = i >> 6, k = i & 63;
        W2T[k*129 + n] = W2[i];
    }
    for (int i = tid; i < 128*128; i += NTHR) {
        int n = i >> 7, k = i & 127;
        WrT[k*129 + n] = Wr[i];
    }
    for (int i = tid; i < 128*128; i += NTHR) {
        int n = i >> 7, k = i & 127;
        W3T[k*129 + n] = W3[i];
    }

    // ---- one-time: W1 B-fragment (32x16 KxN) for this wave's 16 neurons ----
    // B element e of lane half h holds K = 8h + e (e<8), 16 + 8h + (e-8) (e>=8).
    v16h bfrag;
    {
        const int n = wv * 16 + lmod;
        const int kbase = halfsel * 8;
        #pragma unroll
        for (int e = 0; e < 16; ++e)
            bfrag[e] = (_Float16)W1[n * 32 + kbase + e + ((e >= 8) ? 8 : 0)];
    }

    // ---- register-resident LIF state (8 lanes per sample) ----
    const int s   = tid >> 3;
    const int sub = tid & 7;
    float v1[8], v2[16], v3[16];
    #pragma unroll
    for (int k = 0; k < 8;  ++k) v1[k] = 0.f;
    #pragma unroll
    for (int k = 0; k < 16; ++k) { v2[k] = 0.f; v3[k] = 0.f; }
    int j2prev = -1;

    const float DEC = 1.0f - 1.0f / 80000.0f;
    const float VTH = 1.2f;

    for (int t = 0; t < TT; ++t) {
        const int tc = t & (CH - 1);
        __syncthreads();                       // WAR: inp1 / xh from prev iter

        if (tc == 0) {
            // stage + f32->f16 pack: pair p covers input rows l=2p, 2p+1.
            // 16 samples x 16 pairs x 8 float4-segments = 2048 tasks.
            for (int i = tid; i < SPB*16*8; i += NTHR) {
                const int ss  = i >> 7;
                const int rem = i & 127;
                const int p   = rem >> 3;
                const int seg = rem & 7;
                const float* r0 = x + ((size_t)((sb+ss)*32 + 2*p))*TT + t + seg*4;
                const float4 a = *(const float4*)r0;
                const float4 b = *(const float4*)(r0 + TT);
                unsigned* dst = xh + ss*XH_PITCH_M + p*XH_PITCH_P + seg*4;
                dst[0] = pack_h2(a.x, b.x);
                dst[1] = pack_h2(a.y, b.y);
                dst[2] = pack_h2(a.z, b.z);
                dst[3] = pack_h2(a.w, b.w);
            }
            if (t + CH < TT) {                  // prefetch next chunk (near)
                for (int i = tid; i < SPB*32; i += NTHR) {
                    int ss = i >> 5, l = i & 31;
                    __builtin_prefetch(x + ((size_t)((sb+ss)*32 + l))*TT + t + CH, 0, 3);
                }
            }
            __syncthreads();
        }

        // ---- A fragment: 8 x ds_load_b32 of pre-packed half2 pairs ----
        // VGPR r holds K=K0,K0+1 with K0 = 2r + (r>=4?8:0) + (hi-half?8:0)
        union { v16h h; unsigned u[8]; } af;
        #pragma unroll
        for (int r = 0; r < 8; ++r) {
            const int p = r + ((r >= 4) ? 4 : 0) + (halfsel ? 4 : 0);
            af.u[r] = xh[lmod*XH_PITCH_M + p*XH_PITCH_P + tc];
        }
        v8f c = {};
        c = __builtin_amdgcn_wmma_f32_16x16x32_f16(false, af.h, false, bfrag,
                                                   (short)0, c, false, false);
        {   // scatter C: lane -> neuron n, VGPR r -> sample m
            const int n = wv*16 + lmod;
            #pragma unroll
            for (int r = 0; r < 8; ++r)
                inp1[(r + 8*halfsel)*68 + n] = c[r];
        }
        __syncthreads();

        // ---- layer 1: charge, WTA argmax(64), hard reset ----
        float bm = -3.4e38f; int bi = 0;
        #pragma unroll
        for (int k = 0; k < 8; ++k) {
            const int n = sub + 8*k;
            const float vv = v1[k]*DEC + inp1[s*68 + n];
            v1[k] = vv;
            if (vv > bm) { bm = vv; bi = n; }
        }
        const int j1 = wta_spike(bm, bi, VTH);
        #pragma unroll
        for (int k = 0; k < 8; ++k)
            v1[k] = (v1[k] >= VTH) ? 0.f : v1[k];

        // ---- layer 2: spike-sparse drive (branchless gated gathers) ----
        const float g1 = (j1     >= 0) ? 1.f : 0.f;
        const float gr = (j2prev >= 0) ? 1.f : 0.f;
        const int   r1 = (j1     >= 0) ? j1     : 0;
        const int   rr = (j2prev >= 0) ? j2prev : 0;
        bm = -3.4e38f; bi = 0;
        #pragma unroll
        for (int k = 0; k < 16; ++k) {
            const int n = sub + 8*k;
            const float in2 = g1 * W2T[r1*129 + n] + gr * WrT[rr*129 + n];
            const float vv = v2[k]*DEC + in2;
            v2[k] = vv;
            if (vv > bm) { bm = vv; bi = n; }
        }
        const int j2 = wta_spike(bm, bi, VTH);
        #pragma unroll
        for (int k = 0; k < 16; ++k)
            v2[k] = (v2[k] >= VTH) ? 0.f : v2[k];
        j2prev = j2;

        // ---- layer 3: no WTA needed (spikes unused); reset + final pre_v ----
        const float g3 = (j2 >= 0) ? 1.f : 0.f;
        const int   r3 = (j2 >= 0) ? j2  : 0;
        if (t < TT - 1) {
            #pragma unroll
            for (int k = 0; k < 16; ++k) {
                const int n = sub + 8*k;
                const float vv = v3[k]*DEC + g3 * W3T[r3*129 + n];
                v3[k] = (vv >= VTH) ? 0.f : vv;
            }
        } else {
            #pragma unroll
            for (int k = 0; k < 16; ++k) {
                const int n = sub + 8*k;
                const float vv = v3[k]*DEC + g3 * W3T[r3*129 + n]; // pre-reset
                out[(size_t)(sb + s)*128 + n] = expf(vv);
            }
        }
    }
}

extern "C" void kernel_launch(void* const* d_in, const int* in_sizes, int n_in,
                              void* d_out, int out_size, void* d_ws, size_t ws_size,
                              hipStream_t stream) {
    (void)in_sizes; (void)n_in; (void)out_size; (void)d_ws; (void)ws_size;
    const float* x  = (const float*)d_in[0];
    const float* W1 = (const float*)d_in[1];
    const float* W2 = (const float*)d_in[2];
    const float* Wr = (const float*)d_in[3];
    const float* W3 = (const float*)d_in[4];
    float* out = (float*)d_out;

    const size_t shmem = (size_t)LDS_WORDS * sizeof(float); // 205376 B < 320 KB/WGP
    eeg_lsm_scan_kernel<<<dim3(256 / SPB), dim3(NTHR), shmem, stream>>>(
        x, W1, W2, Wr, W3, out);
}
